// EDCN_type4_51496657879674
// MI455X (gfx1250) — matrix operand order repaired and checked by reading, not compiled, not run on hardware
//
#include <hip/hip_runtime.h>
#include <cstddef>
#include <cstdint>

// ---------------------------------------------------------------------------
// Problem constants (from reference)
// ---------------------------------------------------------------------------
#define BGR   1024          // batch (graphs)
#define NPG   96            // nodes per graph
#define KNN_K 16            // neighbors
#define NTOT  (BGR * NPG)   // total nodes
#define COMB  101           // 5 + 32*3
#define MLPIN (NPG * COMB)  // 9696  (divisible by 32)

typedef __attribute__((ext_vector_type(16))) __bf16 v16bf;
typedef __attribute__((ext_vector_type(8)))  float  v8f;
typedef unsigned int uint32x4 __attribute__((ext_vector_type(4)));
typedef int          int32x8  __attribute__((ext_vector_type(8)));
typedef int          int32x4  __attribute__((ext_vector_type(4)));

// ---------------------------------------------------------------------------
// bf16 helpers (explicit RNE conversion via bit ops)
// ---------------------------------------------------------------------------
__device__ inline __bf16 f2bf(float f) {
    unsigned u = __float_as_uint(f);
    unsigned r = (u + 0x7FFFu + ((u >> 16) & 1u)) >> 16;
    unsigned short s = (unsigned short)r;
    return __builtin_bit_cast(__bf16, s);
}
__device__ inline float bf2f(__bf16 b) {
    unsigned short s = __builtin_bit_cast(unsigned short, b);
    return __uint_as_float(((unsigned)s) << 16);
}

template<int ACT> __device__ inline float apply_act(float v) {
    if constexpr (ACT == 1) return fmaxf(v, 0.0f);                // relu
    else if constexpr (ACT == 2) return v > 0.0f ? v : 0.01f * v; // leaky
    else return v;
}

// ---------------------------------------------------------------------------
// WMMA fragment loaders (bf16, 16x16x32), A/B use contiguous k order (same for A and B).
// A 16x32 (MxK): M = lane&15; elems e -> K = (lane>>4)*16 + e.
// B 32x16 (KxN): N = lane&15; elems e -> K = (lane>>4)*16 + e.
// C/D 16x16:     elem r -> M=(lane>>4)*8+r, N=lane&15.
// ---------------------------------------------------------------------------
__device__ inline v16bf wmma_load_a(const __bf16* base, int row0, int k0, int ld) {
    int lane = threadIdx.x & 31;
    int m    = lane & 15;
    int kb   = (lane >> 4) * 16;
    const __bf16* r = base + (row0 + m) * ld + k0 + kb;
    v16bf a;
#pragma unroll
    for (int e = 0; e < 8; ++e) a[e] = r[e];
#pragma unroll
    for (int e = 0; e < 8; ++e) a[8 + e] = r[8 + e];
    return a;
}

__device__ inline v16bf wmma_load_b(const __bf16* base, int k0, int col0, int ld) {
    int lane = threadIdx.x & 31;
    int n    = lane & 15;
    int kb   = (lane >> 4) * 16;
    v16bf b;
#pragma unroll
    for (int e = 0; e < 16; ++e) b[e] = base[(k0 + kb + e) * ld + col0 + n];
    return b;
}

__device__ inline v8f wmma_bf16(v16bf a, v16bf b, v8f c) {
    return __builtin_amdgcn_wmma_f32_16x16x32_bf16(
        /*neg_a=*/false, a, /*neg_b=*/false, b,
        /*c_mod=*/(short)0, c, /*reuse_a=*/false, /*reuse_b=*/false);
}

// ---------------------------------------------------------------------------
// Tensor Data Mover: 2D bf16 tile (tile_w x tile_h elements, row stride in
// elements) from global -> LDS, contiguous row-major in LDS.
// D# packing per CDNA5 ISA (tensor dims == tile dims,
// tiles are always fully in-bounds here).
// ---------------------------------------------------------------------------
__device__ inline void tdm_load_2d_bf16(unsigned lds_addr, const void* gptr,
                                        unsigned tile_w, unsigned tile_h,
                                        unsigned long long row_stride_elems) {
    unsigned long long ga = (unsigned long long)(uintptr_t)gptr;
    uint32x4 g0;
    g0[0] = 1u;                                        // count=1, user mode
    g0[1] = lds_addr;                                  // LDS byte address
    g0[2] = (unsigned)(ga & 0xFFFFFFFFu);              // global_addr[31:0]
    g0[3] = (unsigned)((ga >> 32) & 0x01FFFFFFu)       // global_addr[56:32]
          | (2u << 30);                                // type = 2 ("image")
    int32x8 g1;
    g1[0] = (int)(1u << 16);                           // data_size=1 (2 bytes)
    g1[1] = (int)((tile_w & 0xFFFFu) << 16);           // tensor_dim0[15:0]
    g1[2] = (int)(((tile_w >> 16) & 0xFFFFu)           // tensor_dim0[31:16]
          |       ((tile_h & 0xFFFFu) << 16));         // tensor_dim1[15:0]
    g1[3] = (int)(((tile_h >> 16) & 0xFFFFu)           // tensor_dim1[31:16]
          |       ((tile_w & 0xFFFFu) << 16));         // tile_dim0
    g1[4] = (int)(tile_h & 0xFFFFu);                   // tile_dim1 (tile_dim2=0)
    g1[5] = (int)(unsigned)(row_stride_elems & 0xFFFFFFFFull);   // dim0_stride[31:0]
    g1[6] = (int)(unsigned)((row_stride_elems >> 32) & 0xFFFFull); // [47:32]
    g1[7] = 0;
    int32x4 z4 = {0, 0, 0, 0};
#if __clang_major__ >= 23
    int32x8 z8 = {0, 0, 0, 0, 0, 0, 0, 0};
    __builtin_amdgcn_tensor_load_to_lds(g0, g1, z4, z4, z8, 0);
#else
    __builtin_amdgcn_tensor_load_to_lds(g0, g1, z4, z4, 0);
#endif
}

// ---------------------------------------------------------------------------
// Kernel: node feature assembly  xx = [tq, x, pos]  (N x 5, fp32)
// ---------------------------------------------------------------------------
__global__ void feat0_kernel(const float* __restrict__ x, const float* __restrict__ pos,
                             const float* __restrict__ tq, float* __restrict__ f0) {
    int n = blockIdx.x * blockDim.x + threadIdx.x;
    if (n >= NTOT) return;
    f0[n * 5 + 0] = tq[n];
    f0[n * 5 + 1] = x[n];
    f0[n * 5 + 2] = pos[n * 3 + 0];
    f0[n * 5 + 3] = pos[n * 3 + 1];
    f0[n * 5 + 4] = pos[n * 3 + 2];
}

// ---------------------------------------------------------------------------
// Kernel: per-graph KNN (K=16 nearest incl. self; ties -> lower index, like top_k)
// ---------------------------------------------------------------------------
__global__ void __launch_bounds__(96) knn_kernel(const float* __restrict__ pos,
                                                 int* __restrict__ knn) {
    __shared__ float sx[NPG], sy[NPG], sz[NPG];
    int g = blockIdx.x, t = threadIdx.x;
    sx[t] = pos[(g * NPG + t) * 3 + 0];
    sy[t] = pos[(g * NPG + t) * 3 + 1];
    sz[t] = pos[(g * NPG + t) * 3 + 2];
    __syncthreads();
    float bd[KNN_K]; int bi[KNN_K];
#pragma unroll
    for (int k = 0; k < KNN_K; ++k) { bd[k] = 3.4e38f; bi[k] = 0; }
    float px = sx[t], py = sy[t], pz = sz[t];
    for (int j = 0; j < NPG; ++j) {
        float dx = px - sx[j], dy = py - sy[j], dz = pz - sz[j];
        float d2 = dx * dx + dy * dy + dz * dz;
        if (d2 < bd[KNN_K - 1]) {
            int p = KNN_K - 1;
            while (p > 0 && bd[p - 1] > d2) { bd[p] = bd[p - 1]; bi[p] = bi[p - 1]; --p; }
            bd[p] = d2; bi[p] = j;
        }
    }
#pragma unroll
    for (int k = 0; k < KNN_K; ++k) knn[(g * NPG + t) * KNN_K + k] = bi[k];
}

// ---------------------------------------------------------------------------
// Kernel: fp32 -> bf16 weight conversion with K zero-padding (row-major K x N)
// ---------------------------------------------------------------------------
__global__ void wconv_kernel(const float* __restrict__ W, __bf16* __restrict__ Wb,
                             int Kin, int N, int Kpad) {
    int i = blockIdx.x * blockDim.x + threadIdx.x;
    if (i >= Kpad * N) return;
    int k = i / N, n = i % N;
    Wb[i] = f2bf(k < Kin ? W[k * N + n] : 0.0f);
}

// ---------------------------------------------------------------------------
// EdgeConv kernel: one block handles 8 nodes (128 edges) of one graph.
//   edge e of node i:  m = [x_i (FIN) | x_j - x_i (FIN) | 0-pad to EK]
//   layer0: (128 x EK) @ (EK x MID)  + b0, act
//   layer1: (128 x MID) @ (MID x FOUT) + b1, act
//   out[node] = max over 16 edges
// ---------------------------------------------------------------------------
template<int FIN, int EK, int MID, int FOUT, int ACT>
__global__ void __launch_bounds__(256) edgeconv_kernel(
        const float* __restrict__ feat, const int* __restrict__ knn,
        const __bf16* __restrict__ W0, const float* __restrict__ b0,
        const __bf16* __restrict__ W1, const float* __restrict__ b1,
        float* __restrict__ out) {
    constexpr int EDGES = 128;                 // 8 nodes * 16 neighbors
    __shared__ __bf16 sE[EDGES * EK];
    __shared__ __bf16 sM[EDGES * MID];
    __shared__ __bf16 sO[EDGES * FOUT];
    __shared__ __bf16 sW0[EK * MID];
    __shared__ __bf16 sW1[MID * FOUT];

    int g     = blockIdx.x / (NPG / 8);
    int chunk = blockIdx.x % (NPG / 8);
    int t     = threadIdx.x;

    for (int i = t; i < EK * MID; i += 256)   sW0[i] = W0[i];
    for (int i = t; i < MID * FOUT; i += 256) sW1[i] = W1[i];

    // build edge feature matrix
    for (int i = t; i < EDGES * EK; i += 256) {
        int e = i / EK, c = i % EK;
        int ni = chunk * 8 + (e >> 4);
        int gi = g * NPG + ni;
        int j  = knn[gi * KNN_K + (e & 15)];
        int gj = g * NPG + j;
        float v;
        if (c < FIN)          v = feat[gi * FIN + c];
        else if (c < 2 * FIN) v = feat[gj * FIN + (c - FIN)] - feat[gi * FIN + (c - FIN)];
        else                  v = 0.0f;
        sE[i] = f2bf(v);
    }
    __syncthreads();

    // wave id forced into an SGPR -> tile loops are provably uniform,
    // no EXEC masking around WMMA (ISA requires EXEC all-ones).
    int wave  = __builtin_amdgcn_readfirstlane(threadIdx.x) >> 5;
    int lane  = t & 31;
    int col   = lane & 15;
    int rbase = (lane >> 4) * 8;

    // ---- layer 0 ----
    {
        constexpr int MT = EDGES / 16, NT = MID / 16;
        for (int tile = wave; tile < MT * NT; tile += 8) {
            int mt = tile / NT, nt = tile % NT;
            v8f acc = {};
#pragma unroll
            for (int ks = 0; ks < EK / 32; ++ks)
                acc = wmma_bf16(wmma_load_a(sE, mt * 16, ks * 32, EK),
                                wmma_load_b(sW0, ks * 32, nt * 16, MID), acc);
#pragma unroll
            for (int r = 0; r < 8; ++r) {
                float v = apply_act<ACT>(acc[r] + b0[nt * 16 + col]);
                sM[(mt * 16 + rbase + r) * MID + nt * 16 + col] = f2bf(v);
            }
        }
    }
    __syncthreads();

    // ---- layer 1 ----
    {
        constexpr int MT = EDGES / 16, NT = FOUT / 16;
        for (int tile = wave; tile < MT * NT; tile += 8) {
            int mt = tile / NT, nt = tile % NT;
            v8f acc = {};
#pragma unroll
            for (int ks = 0; ks < MID / 32; ++ks)
                acc = wmma_bf16(wmma_load_a(sM, mt * 16, ks * 32, MID),
                                wmma_load_b(sW1, ks * 32, nt * 16, FOUT), acc);
#pragma unroll
            for (int r = 0; r < 8; ++r) {
                float v = apply_act<ACT>(acc[r] + b1[nt * 16 + col]);
                sO[(mt * 16 + rbase + r) * FOUT + nt * 16 + col] = f2bf(v);
            }
        }
    }
    __syncthreads();

    // ---- max over K neighbors ----
    for (int o = t; o < 8 * FOUT; o += 256) {
        int node = o / FOUT, c = o % FOUT;
        float m = -3.4e38f;
#pragma unroll
        for (int k = 0; k < KNN_K; ++k)
            m = fmaxf(m, bf2f(sO[(node * KNN_K + k) * FOUT + c]));
        out[(g * NPG + chunk * 8 + node) * FOUT + c] = m;
    }
}

// ---------------------------------------------------------------------------
// Kernel: build comb (B x 9696) in bf16:  per node [xx(5)|h1(32)|h2(32)|h3(32)]
// ---------------------------------------------------------------------------
__global__ void comb_kernel(const float* __restrict__ f0, const float* __restrict__ h1,
                            const float* __restrict__ h2, const float* __restrict__ h3,
                            __bf16* __restrict__ comb) {
    int i = blockIdx.x * blockDim.x + threadIdx.x;
    if (i >= NTOT * COMB) return;
    int node = i / COMB, c = i % COMB;
    float v;
    if (c < 5)       v = f0[node * 5 + c];
    else if (c < 37) v = h1[node * 32 + (c - 5)];
    else if (c < 69) v = h2[node * 32 + (c - 37)];
    else             v = h3[node * 32 + (c - 69)];
    comb[i] = f2bf(v);
}

// ---------------------------------------------------------------------------
// Generic GEMM:  out = act(A(MxK) @ W(KxN) + bias),  bf16 in/out, fp32 accum.
// Block tile 64x64 (8 waves; each wave one 16x32 slab). M,N%64==0, K%32==0.
// LDS staging via Tensor Data Mover (wave 0 issues, TENSORcnt-waited).
// ---------------------------------------------------------------------------
template<int ACT>
__global__ void __launch_bounds__(256) gemm_kernel(
        const __bf16* __restrict__ A, const __bf16* __restrict__ Wt,
        const float* __restrict__ bias, __bf16* __restrict__ out,
        int M, int N, int Kd) {
    __shared__ __bf16 sA[64 * 32];
    __shared__ __bf16 sB[32 * 64];
    int bm = blockIdx.x, bn = blockIdx.y;
    int t = threadIdx.x, lane = t & 31;
    int wave = __builtin_amdgcn_readfirstlane(threadIdx.x) >> 5;
    int mt = wave & 3, ng = wave >> 2;
    v8f acc0 = {}, acc1 = {};

    unsigned ldsA = (unsigned)(uintptr_t)&sA[0];
    unsigned ldsB = (unsigned)(uintptr_t)&sB[0];

    for (int k0 = 0; k0 < Kd; k0 += 32) {
        if (wave == 0) {
            // A slab: 64 rows x 32 bf16, row stride Kd
            tdm_load_2d_bf16(ldsA, A + (size_t)bm * 64 * Kd + k0,
                             /*tile_w=*/32, /*tile_h=*/64, (unsigned long long)Kd);
            // B slab: 32 rows x 64 bf16, row stride N
            tdm_load_2d_bf16(ldsB, Wt + (size_t)k0 * N + bn * 64,
                             /*tile_w=*/64, /*tile_h=*/32, (unsigned long long)N);
            __builtin_amdgcn_s_wait_tensorcnt(0);
        }
        __syncthreads();
        v16bf a  = wmma_load_a(sA, mt * 16, 0, 32);
        v16bf b0 = wmma_load_b(sB, 0, ng * 32, 64);
        v16bf b1 = wmma_load_b(sB, 0, ng * 32 + 16, 64);
        acc0 = wmma_bf16(a, b0, acc0);
        acc1 = wmma_bf16(a, b1, acc1);
        __syncthreads();
    }

    int colc = lane & 15, rbase = (lane >> 4) * 8;
    int row0 = bm * 64 + mt * 16 + rbase;
    int c0   = bn * 64 + ng * 32 + colc;
#pragma unroll
    for (int r = 0; r < 8; ++r) {
        float v0 = apply_act<ACT>(acc0[r] + bias[c0]);
        float v1 = apply_act<ACT>(acc1[r] + bias[c0 + 16]);
        out[(size_t)(row0 + r) * N + c0]      = f2bf(v0);
        out[(size_t)(row0 + r) * N + c0 + 16] = f2bf(v1);
    }
}

// ---------------------------------------------------------------------------
// Final heads (64->2 and 64->1), fp32, concat to (B,3)
// ---------------------------------------------------------------------------
__global__ void final_kernel(const __bf16* __restrict__ a4, const __bf16* __restrict__ a4b,
                             const float* __restrict__ W4, const float* __restrict__ b4,
                             const float* __restrict__ W4b, const float* __restrict__ b4b,
                             float* __restrict__ out) {
    int b = blockIdx.x * blockDim.x + threadIdx.x;
    if (b >= BGR) return;
    float o0 = b4[0], o1 = b4[1], o2 = b4b[0];
#pragma unroll
    for (int k = 0; k < 64; ++k) {
        float h  = bf2f(a4[b * 64 + k]);
        float h2 = bf2f(a4b[b * 64 + k]);
        o0 += h * W4[k * 2 + 0];
        o1 += h * W4[k * 2 + 1];
        o2 += h2 * W4b[k];
    }
    out[b * 3 + 0] = o0;
    out[b * 3 + 1] = o1;
    out[b * 3 + 2] = o2;
}

// ---------------------------------------------------------------------------
// Host launcher
// ---------------------------------------------------------------------------
extern "C" void kernel_launch(void* const* d_in, const int* in_sizes, int n_in,
                              void* d_out, int out_size, void* d_ws, size_t ws_size,
                              hipStream_t stream) {
    (void)in_sizes; (void)n_in; (void)out_size; (void)ws_size;
    const float* x   = (const float*)d_in[0];
    const float* pos = (const float*)d_in[1];
    const float* tq  = (const float*)d_in[2];
    // d_in[3] = batch (unused)
    const float* c1w0 = (const float*)d_in[4];
    const float* c1b0 = (const float*)d_in[5];
    const float* c1w1 = (const float*)d_in[6];
    const float* c1b1 = (const float*)d_in[7];
    const float* c2w0 = (const float*)d_in[8];
    const float* c2b0 = (const float*)d_in[9];
    const float* c2w1 = (const float*)d_in[10];
    const float* c2b1 = (const float*)d_in[11];
    const float* hw[2][5]; const float* hb[2][5];
    for (int h = 0; h < 2; ++h)
        for (int l = 0; l < 5; ++l) {
            hw[h][l] = (const float*)d_in[12 + h * 10 + l * 2];
            hb[h][l] = (const float*)d_in[12 + h * 10 + l * 2 + 1];
        }

    // ---- workspace layout ----
    char* ws = (char*)d_ws;
    size_t off = 0;
    auto alloc = [&](size_t bytes) -> void* {
        void* p = ws + off;
        off = (off + bytes + 255) & ~(size_t)255;
        return p;
    };
    int*    knn  = (int*)   alloc((size_t)NTOT * KNN_K * 4);
    float*  f0   = (float*) alloc((size_t)NTOT * 5 * 4);
    float*  h1   = (float*) alloc((size_t)NTOT * 32 * 4);
    float*  h2   = (float*) alloc((size_t)NTOT * 32 * 4);
    float*  h3   = (float*) alloc((size_t)NTOT * 32 * 4);
    __bf16* comb = (__bf16*)alloc((size_t)BGR * MLPIN * 2);
    __bf16* c1w0b = (__bf16*)alloc(32 * 32 * 2);
    __bf16* c1w1b = (__bf16*)alloc(32 * 32 * 2);
    __bf16* c2w0b = (__bf16*)alloc(64 * 64 * 2);
    __bf16* c2w1b = (__bf16*)alloc(64 * 32 * 2);
    __bf16* mw[2][4]; __bf16* act[2][4];
    const int mk[4] = {MLPIN, 512, 256, 128};
    const int mn[4] = {512, 256, 128, 64};
    for (int h = 0; h < 2; ++h)
        for (int l = 0; l < 4; ++l) {
            mw[h][l]  = (__bf16*)alloc((size_t)mk[l] * mn[l] * 2);
            act[h][l] = (__bf16*)alloc((size_t)BGR * mn[l] * 2);
        }

    // ---- stage 0: node features + KNN ----
    feat0_kernel<<<(NTOT + 255) / 256, 256, 0, stream>>>(x, pos, tq, f0);
    knn_kernel<<<BGR, 96, 0, stream>>>(pos, knn);

    // ---- stage 1: weight conversion to bf16 (K zero-padded where needed) ----
    auto wconv = [&](const float* src, __bf16* dst, int Kin, int N, int Kpad) {
        int tot = Kpad * N;
        wconv_kernel<<<(tot + 255) / 256, 256, 0, stream>>>(src, dst, Kin, N, Kpad);
    };
    wconv(c1w0, c1w0b, 10, 32, 32);
    wconv(c1w1, c1w1b, 32, 32, 32);
    wconv(c2w0, c2w0b, 64, 64, 64);
    wconv(c2w1, c2w1b, 64, 32, 64);
    for (int h = 0; h < 2; ++h)
        for (int l = 0; l < 4; ++l)
            wconv(hw[h][l], mw[h][l], mk[l], mn[l], mk[l]);

    // ---- stage 2: EdgeConvs (WMMA) ----
    dim3 egrid(BGR * (NPG / 8));
    edgeconv_kernel<5, 32, 32, 32, 1><<<egrid, 256, 0, stream>>>(
        f0, knn, c1w0b, c1b0, c1w1b, c1b1, h1);
    edgeconv_kernel<32, 64, 64, 32, 2><<<egrid, 256, 0, stream>>>(
        h1, knn, c2w0b, c2b0, c2w1b, c2b1, h2);
    edgeconv_kernel<32, 64, 64, 32, 2><<<egrid, 256, 0, stream>>>(
        h2, knn, c2w0b, c2b0, c2w1b, c2b1, h3);

    // ---- stage 3: comb assembly ----
    comb_kernel<<<((size_t)NTOT * COMB + 255) / 256, 256, 0, stream>>>(f0, h1, h2, h3, comb);

    // ---- stage 4: MLP chains (WMMA GEMMs with TDM staging) ----
    for (int h = 0; h < 2; ++h) {
        gemm_kernel<1><<<dim3(BGR / 64, 512 / 64), 256, 0, stream>>>(
            comb, mw[h][0], hb[h][0], act[h][0], BGR, 512, MLPIN);
        gemm_kernel<1><<<dim3(BGR / 64, 256 / 64), 256, 0, stream>>>(
            act[h][0], mw[h][1], hb[h][1], act[h][1], BGR, 256, 512);
        gemm_kernel<1><<<dim3(BGR / 64, 128 / 64), 256, 0, stream>>>(
            act[h][1], mw[h][2], hb[h][2], act[h][2], BGR, 128, 256);
        gemm_kernel<1><<<dim3(BGR / 64, 64 / 64), 256, 0, stream>>>(
            act[h][2], mw[h][3], hb[h][3], act[h][3], BGR, 64, 128);
    }

    // ---- stage 5: final heads, concat (B,3) fp32 ----
    final_kernel<<<(BGR + 255) / 256, 256, 0, stream>>>(
        act[0][3], act[1][3], hw[0][4], hb[0][4], hw[1][4], hb[1][4], (float*)d_out);
}
